// MultiHeadAttention_25829933318809
// MI455X (gfx1250) — matrix operand order repaired and checked by reading, compiled, not run on hardware
//
#include <hip/hip_runtime.h>

// ---------------------------------------------------------------------------
// MI455X (gfx1250) multi-head attention forward, bf16 WMMA everywhere.
// Software-pipelined operand loads (2-deep rotation) so WMMAs never need a
// full s_wait_loadcnt 0 drain.
// ---------------------------------------------------------------------------

typedef __bf16 bf16;
typedef __attribute__((ext_vector_type(16))) __bf16 v16bf;
typedef __attribute__((ext_vector_type(8)))  __bf16 v8bf;
typedef __attribute__((ext_vector_type(4)))  __bf16 v4bf;
typedef __attribute__((ext_vector_type(8)))  float  v8f;
typedef __attribute__((ext_vector_type(4)))  float  v4f;

union AB { v16bf v; v8bf h[2]; };

__device__ __forceinline__ v8f wmma_bf16(const AB& a, const AB& b, v8f c) {
  return __builtin_amdgcn_wmma_f32_16x16x32_bf16(
      /*neg_a=*/false, a.v, /*neg_b=*/false, b.v,
      /*c_mod=*/(short)0, c, /*reuse_a=*/false, /*reuse_b=*/false);
}

constexpr int Bn = 2, Sn = 2048, En = 768, Hn = 12, Dn = 64;
constexpr int M_TOK = Bn * Sn;           // 4096 tokens

// ---------------------------------------------------------------------------
// f32 -> bf16 elementwise convert, 4 elements/thread
// ---------------------------------------------------------------------------
__global__ __launch_bounds__(256)
void cvt_f32_to_bf16(const float* __restrict__ src, bf16* __restrict__ dst, int n4) {
  int i = blockIdx.x * 256 + threadIdx.x;
  if (i < n4) {
    v4f f = ((const v4f*)src)[i];
    v4bf b;
#pragma unroll
    for (int j = 0; j < 4; ++j) b[j] = (bf16)f[j];
    ((v4bf*)dst)[i] = b;
  }
}

// ---------------------------------------------------------------------------
// QKV projection: [4096 x 768] @ [768 x 2304] (+bias), bf16 WMMA, f32 acc.
// Wave tile 32(M) x 64(N); B operand double-buffered, next k-step's A/B0
// prefetched before the last WMMAs of the current step.
// Writes Q (pre-scaled by 1/sqrt(D)) and K as [B,H,S,D] bf16, V transposed
// as [B,H,D,S] bf16.
// ---------------------------------------------------------------------------
__global__ __launch_bounds__(256, 4)
void qkv_gemm(const bf16* __restrict__ xb, const bf16* __restrict__ wb,
              const float* __restrict__ bias,
              bf16* __restrict__ qo, bf16* __restrict__ ko, bf16* __restrict__ vto)
{
  const int lane = threadIdx.x & 31;
  const int wave = threadIdx.x >> 5;
  const int gid  = blockIdx.x * 8 + wave;
  const int NT   = (3 * En) / 64;            // 36 N-tiles
  const int mt   = gid / NT;
  const int nt   = gid % NT;
  const int l16  = lane & 15;
  const int kh   = lane >> 4;
  const int row0 = mt * 32, col0 = nt * 64;

  v8f acc[2][4] = {};
  const bf16* arow0 = xb + (size_t)(row0 +      l16) * En;
  const bf16* arow1 = xb + (size_t)(row0 + 16 + l16) * En;

  auto loadA = [&](AB* d, int k0) {
    d[0].h[0] = *(const v8bf*)(arow0 + k0 +      kh * 8);
    d[0].h[1] = *(const v8bf*)(arow0 + k0 + 16 + kh * 8);
    d[1].h[0] = *(const v8bf*)(arow1 + k0 +      kh * 8);
    d[1].h[1] = *(const v8bf*)(arow1 + k0 + 16 + kh * 8);
  };
  auto loadB = [&](AB& d, int k0, int t) {
    const bf16* bcol = wb + (size_t)(col0 + t * 16 + l16) * En + k0 + kh * 8;
    d.h[0] = *(const v8bf*)(bcol);
    d.h[1] = *(const v8bf*)(bcol + 16);
  };

  AB a[2], an[2], b0, b1;
  loadA(a, 0);
  loadB(b0, 0, 0);
  for (int k0 = 0; k0 < En; k0 += 32) {
    loadB(b1, k0, 1);
    acc[0][0] = wmma_bf16(a[0], b0, acc[0][0]);
    acc[1][0] = wmma_bf16(a[1], b0, acc[1][0]);
    loadB(b0, k0, 2);
    acc[0][1] = wmma_bf16(a[0], b1, acc[0][1]);
    acc[1][1] = wmma_bf16(a[1], b1, acc[1][1]);
    loadB(b1, k0, 3);
    acc[0][2] = wmma_bf16(a[0], b0, acc[0][2]);
    acc[1][2] = wmma_bf16(a[1], b0, acc[1][2]);
    const int kn = k0 + 32;
    if (kn < En) { loadA(an, kn); loadB(b0, kn, 0); }
    acc[0][3] = wmma_bf16(a[0], b1, acc[0][3]);
    acc[1][3] = wmma_bf16(a[1], b1, acc[1][3]);
    a[0] = an[0]; a[1] = an[1];
  }

  const int qkv_id = col0 / En;              // 0=Q, 1=K, 2=V
  const int eoff   = col0 % En;
  const int head   = eoff / Dn;
  const int b_     = row0 / Sn;
  const int bh     = b_ * Hn + head;
  const float qscale = 0.125f;               // D^-0.5, folded into Q

#pragma unroll
  for (int m = 0; m < 2; ++m) {
    const int s0 = (row0 % Sn) + m * 16 + kh * 8;
#pragma unroll
    for (int t = 0; t < 4; ++t) {
      const int d  = t * 16 + l16;
      const float bv = bias[col0 + t * 16 + l16];
      if (qkv_id == 0) {
        bf16* p = qo + ((size_t)bh * Sn + s0) * Dn + d;
#pragma unroll
        for (int r = 0; r < 8; ++r) p[(size_t)r * Dn] = (bf16)((acc[m][t][r] + bv) * qscale);
      } else if (qkv_id == 1) {
        bf16* p = ko + ((size_t)bh * Sn + s0) * Dn + d;
#pragma unroll
        for (int r = 0; r < 8; ++r) p[(size_t)r * Dn] = (bf16)(acc[m][t][r] + bv);
      } else {
        v8bf pv;
#pragma unroll
        for (int r = 0; r < 8; ++r) pv[r] = (bf16)(acc[m][t][r] + bv);
        *(v8bf*)(vto + ((size_t)bh * Dn + d) * Sn + s0) = pv;   // contiguous in s
      }
    }
  }
}

// ---------------------------------------------------------------------------
// Flash attention, causal. Block = 8 waves = 128 query rows of one (b,h).
// Wave: 16 query rows; key blocks of 64; online softmax; P re-layout C->A via
// private padded LDS tile. Only the last (diagonal-straddling) key block is
// masked (valid since q0 % 16 == 0). K/V tiles double-buffered; the first V
// tiles are issued BEFORE the softmax so their latency hides under the VALU.
// ---------------------------------------------------------------------------
__global__ __launch_bounds__(256, 4)
void attn_fa(const bf16* __restrict__ q, const bf16* __restrict__ k,
             const bf16* __restrict__ vt, bf16* __restrict__ ao)
{
  __shared__ __align__(16) bf16 lds[8][16][72];   // 72 = 64 + pad (bank spread)

  const int lane = threadIdx.x & 31;
  const int wave = threadIdx.x >> 5;
  const int l16  = lane & 15;
  const int kh   = lane >> 4;
  const int QB   = Sn / 128;                 // 16 query blocks per (b,h)
  const int bh   = blockIdx.x / QB;
  const int q0   = (blockIdx.x % QB) * 128 + wave * 16;

  // Q in A-layout: two K=32 slices over D=64
  const bf16* qrow = q + ((size_t)bh * Sn + q0 + l16) * Dn;
  AB qa[2];
  qa[0].h[0] = *(const v8bf*)(qrow +      kh * 8);
  qa[0].h[1] = *(const v8bf*)(qrow + 16 + kh * 8);
  qa[1].h[0] = *(const v8bf*)(qrow + 32 + kh * 8);
  qa[1].h[1] = *(const v8bf*)(qrow + 48 + kh * 8);

  v8f o[4] = {};
  float mrow[8], lrow[8];
#pragma unroll
  for (int r = 0; r < 8; ++r) { mrow[r] = -1e30f; lrow[r] = 0.f; }

  const int nkb = (q0 + 15) / 64 + 1;        // causal key-block count
  for (int kb = 0; kb < nkb; ++kb) {
    const int kbase = kb * 64;
    const bool diag = (kb == nkb - 1);       // wave-uniform branch

    auto loadK = [&](AB* d, int t) {
      const bf16* krow = k + ((size_t)bh * Sn + kbase + t * 16 + l16) * Dn;
      d[0].h[0] = *(const v8bf*)(krow + kh * 8);
      d[0].h[1] = *(const v8bf*)(krow + 16 + kh * 8);
      d[1].h[0] = *(const v8bf*)(krow + 32 + kh * 8);
      d[1].h[1] = *(const v8bf*)(krow + 48 + kh * 8);
    };
    auto loadV = [&](AB* d, int t) {
      const bf16* vrow = vt + ((size_t)bh * Dn + t * 16 + l16) * Sn + kbase;
      d[0].h[0] = *(const v8bf*)(vrow + kh * 8);
      d[0].h[1] = *(const v8bf*)(vrow + 16 + kh * 8);
      d[1].h[0] = *(const v8bf*)(vrow + 32 + kh * 8);
      d[1].h[1] = *(const v8bf*)(vrow + 48 + kh * 8);
    };

    // ---- S = Q @ K^T : K tiles double-buffered ----
    v8f sacc[4] = {};
    AB kA[2], kB[2];
    loadK(kA, 0);
    loadK(kB, 1);
    sacc[0] = wmma_bf16(qa[0], kA[0], sacc[0]);
    sacc[0] = wmma_bf16(qa[1], kA[1], sacc[0]);
    loadK(kA, 2);
    sacc[1] = wmma_bf16(qa[0], kB[0], sacc[1]);
    sacc[1] = wmma_bf16(qa[1], kB[1], sacc[1]);
    loadK(kB, 3);
    sacc[2] = wmma_bf16(qa[0], kA[0], sacc[2]);
    sacc[2] = wmma_bf16(qa[1], kA[1], sacc[2]);
    sacc[3] = wmma_bf16(qa[0], kB[0], sacc[3]);
    sacc[3] = wmma_bf16(qa[1], kB[1], sacc[3]);

    // ---- V tiles 0/1 issued now: latency hides under the softmax VALU ----
    AB vA[2], vB[2];
    loadV(vA, 0);
    loadV(vB, 1);

    // ---- causal mask (diagonal block only) + row max ----
    float mnew[8];
    if (diag) {
#pragma unroll
      for (int r = 0; r < 8; ++r) {
        const int qidx = q0 + kh * 8 + r;
        float mx = -1e30f;
#pragma unroll
        for (int t = 0; t < 4; ++t) {
          const int kidx = kbase + t * 16 + l16;
          float sv = (kidx <= qidx) ? sacc[t][r] : -1e30f;
          sacc[t][r] = sv;
          mx = fmaxf(mx, sv);
        }
        mnew[r] = mx;
      }
    } else {
#pragma unroll
      for (int r = 0; r < 8; ++r) {
        float mx = sacc[0][r];
#pragma unroll
        for (int t = 1; t < 4; ++t) mx = fmaxf(mx, sacc[t][r]);
        mnew[r] = mx;
      }
    }
#pragma unroll
    for (int off = 1; off < 16; off <<= 1)
#pragma unroll
      for (int r = 0; r < 8; ++r)
        mnew[r] = fmaxf(mnew[r], __shfl_xor(mnew[r], off, 32));

    float alpha[8];
#pragma unroll
    for (int r = 0; r < 8; ++r) {
      mnew[r]  = fmaxf(mnew[r], mrow[r]);
      alpha[r] = __expf(mrow[r] - mnew[r]);
      mrow[r]  = mnew[r];
    }

    // ---- P = exp(S - m), row sums ----
    float ls[8];
#pragma unroll
    for (int r = 0; r < 8; ++r) ls[r] = 0.f;
#pragma unroll
    for (int t = 0; t < 4; ++t)
#pragma unroll
      for (int r = 0; r < 8; ++r) {
        float p = __expf(sacc[t][r] - mrow[r]);
        sacc[t][r] = p;
        ls[r] += p;
      }
#pragma unroll
    for (int off = 1; off < 16; off <<= 1)
#pragma unroll
      for (int r = 0; r < 8; ++r)
        ls[r] += __shfl_xor(ls[r], off, 32);
#pragma unroll
    for (int r = 0; r < 8; ++r) lrow[r] = lrow[r] * alpha[r] + ls[r];
#pragma unroll
    for (int t = 0; t < 4; ++t)
#pragma unroll
      for (int r = 0; r < 8; ++r) o[t][r] *= alpha[r];

    // ---- re-layout P: C-layout -> A-layout via per-wave LDS tile ----
#pragma unroll
    for (int t = 0; t < 4; ++t)
#pragma unroll
      for (int r = 0; r < 8; ++r)
        lds[wave][kh * 8 + r][t * 16 + l16] = (bf16)sacc[t][r];
    asm volatile("s_wait_dscnt 0" ::: "memory");   // per-wave LDS ordering

    AB pa[2];
    pa[0].h[0] = *(const v8bf*)&lds[wave][l16][     kh * 8];
    pa[0].h[1] = *(const v8bf*)&lds[wave][l16][16 + kh * 8];
    pa[1].h[0] = *(const v8bf*)&lds[wave][l16][32 + kh * 8];
    pa[1].h[1] = *(const v8bf*)&lds[wave][l16][48 + kh * 8];

    // ---- O += P @ V : V tiles double-buffered ----
    o[0] = wmma_bf16(pa[0], vA[0], o[0]);
    o[0] = wmma_bf16(pa[1], vA[1], o[0]);
    loadV(vA, 2);
    o[1] = wmma_bf16(pa[0], vB[0], o[1]);
    o[1] = wmma_bf16(pa[1], vB[1], o[1]);
    loadV(vB, 3);
    o[2] = wmma_bf16(pa[0], vA[0], o[2]);
    o[2] = wmma_bf16(pa[1], vA[1], o[2]);
    o[3] = wmma_bf16(pa[0], vB[0], o[3]);
    o[3] = wmma_bf16(pa[1], vB[1], o[3]);
  }

  // ---- epilogue: normalize, store bf16 to [B,S,E] (E = H*D) ----
  const int b_ = bh / Hn, head = bh % Hn;
#pragma unroll
  for (int t = 0; t < 4; ++t)
#pragma unroll
    for (int r = 0; r < 8; ++r) {
      const int s = q0 + kh * 8 + r;
      ao[((size_t)b_ * Sn + s) * En + head * Dn + t * 16 + l16] =
          (bf16)(o[t][r] / lrow[r]);
    }
}

// ---------------------------------------------------------------------------
// Output projection: [4096 x 768] @ [768 x 768] (+bias) -> f32 out.
// 32(M) x 64(N) wave tile, B double-buffered, next k-step prefetched.
// ---------------------------------------------------------------------------
__global__ __launch_bounds__(256, 4)
void out_gemm(const bf16* __restrict__ ab, const bf16* __restrict__ wb,
              const float* __restrict__ bias, float* __restrict__ out)
{
  const int lane = threadIdx.x & 31;
  const int wave = threadIdx.x >> 5;
  const int gid  = blockIdx.x * 8 + wave;
  const int NT   = En / 64;                  // 12
  const int mt   = gid / NT;
  const int nt   = gid % NT;
  const int l16  = lane & 15;
  const int kh   = lane >> 4;
  const int row0 = mt * 32, col0 = nt * 64;

  v8f acc[2][4] = {};
  const bf16* arow0 = ab + (size_t)(row0 +      l16) * En;
  const bf16* arow1 = ab + (size_t)(row0 + 16 + l16) * En;

  auto loadA = [&](AB* d, int k0) {
    d[0].h[0] = *(const v8bf*)(arow0 + k0 +      kh * 8);
    d[0].h[1] = *(const v8bf*)(arow0 + k0 + 16 + kh * 8);
    d[1].h[0] = *(const v8bf*)(arow1 + k0 +      kh * 8);
    d[1].h[1] = *(const v8bf*)(arow1 + k0 + 16 + kh * 8);
  };
  auto loadB = [&](AB& d, int k0, int t) {
    const bf16* bcol = wb + (size_t)(col0 + t * 16 + l16) * En + k0 + kh * 8;
    d.h[0] = *(const v8bf*)(bcol);
    d.h[1] = *(const v8bf*)(bcol + 16);
  };

  AB a[2], an[2], b0, b1;
  loadA(a, 0);
  loadB(b0, 0, 0);
  for (int k0 = 0; k0 < En; k0 += 32) {
    loadB(b1, k0, 1);
    acc[0][0] = wmma_bf16(a[0], b0, acc[0][0]);
    acc[1][0] = wmma_bf16(a[1], b0, acc[1][0]);
    loadB(b0, k0, 2);
    acc[0][1] = wmma_bf16(a[0], b1, acc[0][1]);
    acc[1][1] = wmma_bf16(a[1], b1, acc[1][1]);
    loadB(b1, k0, 3);
    acc[0][2] = wmma_bf16(a[0], b0, acc[0][2]);
    acc[1][2] = wmma_bf16(a[1], b0, acc[1][2]);
    const int kn = k0 + 32;
    if (kn < En) { loadA(an, kn); loadB(b0, kn, 0); }
    acc[0][3] = wmma_bf16(a[0], b1, acc[0][3]);
    acc[1][3] = wmma_bf16(a[1], b1, acc[1][3]);
    a[0] = an[0]; a[1] = an[1];
  }

#pragma unroll
  for (int m = 0; m < 2; ++m)
#pragma unroll
    for (int t = 0; t < 4; ++t) {
      const float bv = bias[col0 + t * 16 + l16];
#pragma unroll
      for (int r = 0; r < 8; ++r)
        out[(size_t)(row0 + m * 16 + kh * 8 + r) * En + col0 + t * 16 + l16] =
            acc[m][t][r] + bv;
    }
}

// ---------------------------------------------------------------------------
// Host launcher
// ---------------------------------------------------------------------------
extern "C" void kernel_launch(void* const* d_in, const int* in_sizes, int n_in,
                              void* d_out, int out_size, void* d_ws, size_t ws_size,
                              hipStream_t stream)
{
  (void)in_sizes; (void)n_in; (void)out_size; (void)ws_size;
  const float* x     = (const float*)d_in[0];
  // d_in[1] = causal tril mask -> implemented analytically (kidx <= qidx)
  const float* w_qkv = (const float*)d_in[2];
  const float* b_qkv = (const float*)d_in[3];
  const float* w_out = (const float*)d_in[4];
  const float* b_out = (const float*)d_in[5];
  float* out = (float*)d_out;

  char* ws = (char*)d_ws;
  auto carve = [&](size_t bytes) -> char* {
    char* p = ws;
    ws += (bytes + 255) & ~size_t(255);
    return p;
  };
  bf16* xb   = (bf16*)carve((size_t)M_TOK * En * 2);          // x in bf16
  bf16* wqb  = (bf16*)carve((size_t)3 * En * En * 2);         // w_qkv bf16
  bf16* wob  = (bf16*)carve((size_t)En * En * 2);             // w_out bf16
  bf16* qws  = (bf16*)carve((size_t)Bn * Hn * Sn * Dn * 2);   // Q [B,H,S,D]
  bf16* kws  = (bf16*)carve((size_t)Bn * Hn * Sn * Dn * 2);   // K [B,H,S,D]
  bf16* vtws = (bf16*)carve((size_t)Bn * Hn * Dn * Sn * 2);   // V^T [B,H,D,S]
  bf16* aws  = (bf16*)carve((size_t)M_TOK * En * 2);          // attn out bf16

  const int nx4 = M_TOK * En / 4, nwq4 = 3 * En * En / 4, nwo4 = En * En / 4;
  cvt_f32_to_bf16<<<(nx4  + 255) / 256, 256, 0, stream>>>(x,     xb,  nx4);
  cvt_f32_to_bf16<<<(nwq4 + 255) / 256, 256, 0, stream>>>(w_qkv, wqb, nwq4);
  cvt_f32_to_bf16<<<(nwo4 + 255) / 256, 256, 0, stream>>>(w_out, wob, nwo4);

  qkv_gemm<<<(M_TOK / 32) * ((3 * En) / 64) / 8, 256, 0, stream>>>(
      xb, wqb, b_qkv, qws, kws, vtws);

  attn_fa<<<Bn * Hn * (Sn / 128), 256, 0, stream>>>(qws, kws, vtws, aws);

  out_gemm<<<(M_TOK / 32) * (En / 64) / 8, 256, 0, stream>>>(
      aws, wob, b_out, out);
}
